// GeometricAttention_19774029431760
// MI455X (gfx1250) — matrix operand order repaired and checked by reading, not compiled, not run on hardware
//
#include <hip/hip_runtime.h>

// ---------------------------------------------------------------------------
// GeometricAttention on MI455X (gfx1250, wave32, WMMA bf16 16x16x32)
//
// out = normalize(exp(-cdist(x, positions)/(|T|+0.1))) @ values
// Flash-style: per 16-row tile, stream N in chunks of 32, WMMA both GEMMs.
// ---------------------------------------------------------------------------

typedef __bf16 bf16x16 __attribute__((ext_vector_type(16)));
typedef float  f32x8   __attribute__((ext_vector_type(8)));

union V16Q { bf16x16 v; uint4 q[2]; };          // built from two b128 loads
union V16U { bf16x16 v; unsigned int u[8]; };   // built from eight b32 LDS loads

__device__ __forceinline__ unsigned short f2bf(float f) {
    union { float f; unsigned int u; } c; c.f = f;
    unsigned int u = c.u + 0x7FFFu + ((c.u >> 16) & 1u);   // round-to-nearest-even
    return (unsigned short)(u >> 16);
}

// ---- prep: f32 rows -> bf16 rows + per-row squared norm (one wave per row) --
__global__ void prep_rows(const float* __restrict__ src,
                          unsigned short* __restrict__ dst,
                          float* __restrict__ nrm, int rows) {
    const int lane = threadIdx.x & 31;
    const int wave = threadIdx.x >> 5;
    const int row  = blockIdx.x * (blockDim.x >> 5) + wave;
    if (row >= rows) return;
    const float4 v = ((const float4*)(src + (size_t)row * 128))[lane];
    float ss = v.x * v.x + v.y * v.y + v.z * v.z + v.w * v.w;
    uint2 o;
    o.x = (unsigned)f2bf(v.x) | ((unsigned)f2bf(v.y) << 16);
    o.y = (unsigned)f2bf(v.z) | ((unsigned)f2bf(v.w) << 16);
    ((uint2*)(dst + (size_t)row * 128))[lane] = o;
#pragma unroll
    for (int m = 16; m >= 1; m >>= 1) ss += __shfl_xor(ss, m, 32);
    if (lane == 0) nrm[row] = ss;
}

// ---- prep: values[N][128] -> bf16 values_t[128][N] -------------------------
__global__ void prep_vt(const float* __restrict__ vals,
                        unsigned short* __restrict__ vt, int N) {
    const int idx = blockIdx.x * blockDim.x + threadIdx.x;
    if (idx >= N * 128) return;
    const int n = idx % N;
    const int d = idx / N;
    vt[(size_t)d * N + n] = f2bf(vals[(size_t)n * 128 + d]);
}

// ---- prep: inverse temperature --------------------------------------------
__global__ void prep_temp(const float* __restrict__ t,
                          float* __restrict__ it, int N) {
    const int i = blockIdx.x * blockDim.x + threadIdx.x;
    if (i < N) it[i] = 1.0f / (fabsf(t[i]) + 0.1f);
}

// ---- main: one wave owns 16 output rows, streams all N ---------------------
__global__ __launch_bounds__(128)
void geo_attn_main(const unsigned short* __restrict__ xb,  // [M][128] bf16
                   const float* __restrict__ xn,           // [M] |x|^2
                   const unsigned short* __restrict__ pb,  // [N][128] bf16
                   const float* __restrict__ pn,           // [N] |p|^2
                   const float* __restrict__ it,           // [N] 1/(|T|+.1)
                   const unsigned short* __restrict__ vt,  // [128][N] bf16
                   float* __restrict__ out,                // [M][128]
                   int numTiles, int N) {
    __shared__ unsigned short Wlds[4][16][34];  // padded stride: bank-spread

    const int lane = threadIdx.x & 31;
    const int wave = threadIdx.x >> 5;
    const int tile = blockIdx.x * 4 + wave;
    if (tile >= numTiles) return;              // wave-uniform: EXEC stays full
    const int m0   = tile * 16;
    const int col  = lane & 15;
    const int half = lane >> 4;

    // A operand (x tile, 16x128 bf16) in WMMA A-layout: resident all loop long
    V16Q Ax[4];
    const uint4* xrow = (const uint4*)(xb + (size_t)(m0 + col) * 128);
#pragma unroll
    for (int kc = 0; kc < 4; ++kc) {
        Ax[kc].q[0] = xrow[kc * 4 + half];      // k = kc*32 + 8h .. +7
        Ax[kc].q[1] = xrow[kc * 4 + half + 2];  // k = kc*32 + 16 + 8h ..
    }

    float xnr[8];
#pragma unroll
    for (int r = 0; r < 8; ++r) xnr[r] = xn[m0 + r + 8 * half];

    f32x8 acc[8];
#pragma unroll
    for (int d = 0; d < 8; ++d) acc[d] = (f32x8){0.f,0.f,0.f,0.f,0.f,0.f,0.f,0.f};
    float wsum[8] = {0.f,0.f,0.f,0.f,0.f,0.f,0.f,0.f};

    for (int n0 = 0; n0 < N; n0 += 32) {
        f32x8 s2[2];
#pragma unroll
        for (int sub = 0; sub < 2; ++sub) {
            const int nb = n0 + sub * 16 + col;
            const uint4* prow = (const uint4*)(pb + (size_t)nb * 128);
            f32x8 sc = (f32x8){0.f,0.f,0.f,0.f,0.f,0.f,0.f,0.f};
#pragma unroll
            for (int kc = 0; kc < 4; ++kc) {    // S += x_chunk @ p_chunk^T
                V16Q Bp;
                Bp.q[0] = prow[kc * 4 + half];          // k = kc*32 + 8h ..
                Bp.q[1] = prow[kc * 4 + half + 2];
                sc = __builtin_amdgcn_wmma_f32_16x16x32_bf16(
                        false, Ax[kc].v, false, Bp.v, (short)0, sc, false, false);
            }
            const float pnl = pn[nb];
            const float itl = it[nb];
#pragma unroll
            for (int r = 0; r < 8; ++r) {       // sq -> dist -> weight
                float sq = xnr[r] + pnl - 2.0f * sc[r];
                float dd = sqrtf(fmaxf(sq, 0.0f));
                float w  = __expf(-dd * itl);
                wsum[r] += w;
                sc[r] = w;
            }
            s2[sub] = sc;
        }

        // Re-lay 16x32 weight tile from C-layout into A-layout (wave-private LDS)
#pragma unroll
        for (int r = 0; r < 8; ++r) {
            Wlds[wave][r + 8 * half][col]      = f2bf(s2[0][r]);
            Wlds[wave][r + 8 * half][16 + col] = f2bf(s2[1][r]);
        }
        __asm__ volatile("s_wait_dscnt 0" ::: "memory");  // DS in-order per wave
        V16U W;
#pragma unroll
        for (int v = 0; v < 8; ++v) {
            const int k0 = 16 * (v >> 2) + 2 * (v & 3) + 8 * half;
            W.u[v] = *(const unsigned int*)&Wlds[wave][col][k0];
        }
        __asm__ volatile("" ::: "memory");      // keep loads before next stores

        // out_tile += W(16x32) @ values(32x128), 8 WMMAs across D
#pragma unroll
        for (int dt = 0; dt < 8; ++dt) {
            const uint4* vrow = (const uint4*)(vt + (size_t)(dt * 16 + col) * N + n0);
            V16Q Vb;
            Vb.q[0] = vrow[half];               // k(=n_local) = 8h .. +7, 16+8h .. +7
            Vb.q[1] = vrow[half + 2];
            acc[dt] = __builtin_amdgcn_wmma_f32_16x16x32_bf16(
                        false, W.v, false, Vb.v, (short)0, acc[dt], false, false);
        }
    }

    // row-sum of weights: reduce n across the 16-lane half (xor <= 8 stays in half)
#pragma unroll
    for (int r = 0; r < 8; ++r) {
        float s = wsum[r];
#pragma unroll
        for (int m = 8; m >= 1; m >>= 1) s += __shfl_xor(s, m, 32);
        wsum[r] = s + 1e-8f;
    }

#pragma unroll
    for (int dt = 0; dt < 8; ++dt)
#pragma unroll
        for (int r = 0; r < 8; ++r)
            out[(size_t)(m0 + r + 8 * half) * 128 + dt * 16 + col] =
                acc[dt][r] / wsum[r];
}

// ---------------------------------------------------------------------------
extern "C" void kernel_launch(void* const* d_in, const int* in_sizes, int n_in,
                              void* d_out, int out_size, void* d_ws, size_t ws_size,
                              hipStream_t stream) {
    const float* x    = (const float*)d_in[0];
    const float* pos  = (const float*)d_in[1];
    const float* vals = (const float*)d_in[2];
    const float* temp = (const float*)d_in[3];
    const int D = 128;
    const int M = in_sizes[0] / D;   // 16384
    const int N = in_sizes[1] / D;   // 4096

    char* ws = (char*)d_ws;
    size_t off = 0;
    auto take = [&](size_t bytes) {
        size_t p = off;
        off = (off + bytes + 255) & ~(size_t)255;
        return p;
    };
    unsigned short* xb = (unsigned short*)(ws + take((size_t)M * D * 2));
    unsigned short* pb = (unsigned short*)(ws + take((size_t)N * D * 2));
    unsigned short* vt = (unsigned short*)(ws + take((size_t)N * D * 2));
    float* xn = (float*)(ws + take((size_t)M * 4));
    float* pn = (float*)(ws + take((size_t)N * 4));
    float* it = (float*)(ws + take((size_t)N * 4));

    prep_rows<<<(M + 7) / 8, 256, 0, stream>>>(x,   xb, xn, M);
    prep_rows<<<(N + 7) / 8, 256, 0, stream>>>(pos, pb, pn, N);
    prep_vt  <<<(N * D + 255) / 256, 256, 0, stream>>>(vals, vt, N);
    prep_temp<<<(N + 255) / 256,     256, 0, stream>>>(temp, it, N);

    const int numTiles = M / 16;                 // 1024
    const int blocks   = (numTiles + 3) / 4;     // 4 waves / block
    geo_attn_main<<<blocks, 128, 0, stream>>>(xb, xn, pb, pn, it, vt,
                                              (float*)d_out, numTiles, N);
}
